// SVAE_LDS_81355270521045
// MI455X (gfx1250) — matrix operand order repaired and checked by reading, not compiled, not run on hardware
//
#include <hip/hip_runtime.h>
#include <math.h>

#define Tn 32768
#define Dn 1024
#define Hn 6
#define W1K 1032        // 1024 + 8 halves pad: row stride % 64 banks -> conflict-free
#define W2K 40          // 32 + 8 halves pad: row stride 20 dwords mod 64 -> conflict-free

typedef __attribute__((ext_vector_type(16))) _Float16 v16h;
typedef __attribute__((ext_vector_type(8)))  float    v8f;

// ---------- small 4x4 helpers (fully unrolled -> registers) ----------
__device__ __forceinline__ void chol4(const float* S, float* L, float* di) {
  #pragma unroll
  for (int j = 0; j < 4; ++j) {
    float d = S[j*4+j];
    #pragma unroll
    for (int k = 0; k < 4; ++k) if (k < j) d -= L[j*4+k]*L[j*4+k];
    float lj = sqrtf(d);
    L[j*4+j] = lj;
    di[j] = 1.0f / lj;
    #pragma unroll
    for (int i = 0; i < 4; ++i) if (i > j) {
      float s = S[i*4+j];
      #pragma unroll
      for (int k = 0; k < 4; ++k) if (k < j) s -= L[i*4+k]*L[j*4+k];
      L[i*4+j] = s * di[j];
    }
  }
}

// Solves (L L^T) X = B (4x4 RHS), stores G = X^T  (G[c][i] = X[i][c])
__device__ __forceinline__ void cho_solve4_T(const float* L, const float* di,
                                             const float* B, float* G) {
  #pragma unroll
  for (int c = 0; c < 4; ++c) {
    float yv[4], xv[4];
    #pragma unroll
    for (int i = 0; i < 4; ++i) {
      float s = B[i*4+c];
      #pragma unroll
      for (int k = 0; k < 4; ++k) if (k < i) s -= L[i*4+k]*yv[k];
      yv[i] = s * di[i];
    }
    #pragma unroll
    for (int i = 3; i >= 0; --i) {
      float s = yv[i];
      #pragma unroll
      for (int k = 0; k < 4; ++k) if (k > i) s -= L[k*4+i]*xv[k];
      xv[i] = s * di[i];
    }
    #pragma unroll
    for (int i = 0; i < 4; ++i) G[c*4+i] = xv[i];
  }
}

__device__ __forceinline__ void mm4(const float* A, const float* B, float* C) {
  #pragma unroll
  for (int i = 0; i < 4; ++i)
    #pragma unroll
    for (int j = 0; j < 4; ++j) {
      float s = 0.f;
      #pragma unroll
      for (int k = 0; k < 4; ++k) s += A[i*4+k]*B[k*4+j];
      C[i*4+j] = s;
    }
}

// C = A * B^T
__device__ __forceinline__ void mm4_bt(const float* A, const float* B, float* C) {
  #pragma unroll
  for (int i = 0; i < 4; ++i)
    #pragma unroll
    for (int j = 0; j < 4; ++j) {
      float s = 0.f;
      #pragma unroll
      for (int k = 0; k < 4; ++k) s += A[i*4+k]*B[j*4+k];
      C[i*4+j] = s;
    }
}

// ---------------- Encoder: h = relu(xW1+b1); mu, sigma -------------------
__global__ __launch_bounds__(128) void enc_kernel(
    const float* __restrict__ x,  const float* __restrict__ W1, const float* __restrict__ b1,
    const float* __restrict__ Wmu, const float* __restrict__ bmu,
    const float* __restrict__ Wlv, const float* __restrict__ blv,
    float* __restrict__ mu_out, float* __restrict__ sig_out)
{
  // [n][k] f16 image of W1, N padded 6->16 with zeros so the B-fragment build
  // is branch-free for all 32 lanes; row stride 1032 avoids bank conflicts.
  __shared__ _Float16 w1t[16][W1K];
  __shared__ float    hst[4][16][8];     // per-wave h tile staging
  const int tid = threadIdx.x;
  for (int idx = tid; idx < 16*W1K; idx += 128) {
    int n = idx / W1K, k = idx % W1K;
    w1t[n][k] = (_Float16)((n < Hn && k < Dn) ? W1[(size_t)k*Hn + n] : 0.f);
  }
  __syncthreads();

  const int wave = tid >> 5, lane = tid & 31;
  const int m0  = (blockIdx.x*4 + wave) * 16;
  const int hi  = lane >> 4;             // lane-half select
  const int n   = lane & 15;
  const int row = m0 + n;

  v8f acc = {};
  for (int k0 = 0; k0 < Dn; k0 += 32) {
    // A fragment: M=lane&15, K halves {k0+8hi .. +7} and {+16 ..}
    const float* xp = x + (size_t)row*Dn + k0 + hi*8;
    v16h a;
    #pragma unroll
    for (int i = 0; i < 8; ++i) a[i]   = (_Float16)xp[i];
    #pragma unroll
    for (int i = 0; i < 8; ++i) a[8+i] = (_Float16)xp[16+i];
    // B fragment: N=lane&15, K halves {k0+8hi .. +7} and {+16 ..} (branch-free)
    const _Float16* bp = &w1t[n][k0 + hi*8];
    v16h b;
    #pragma unroll
    for (int i = 0; i < 8; ++i) b[i]   = bp[i];
    #pragma unroll
    for (int i = 0; i < 8; ++i) b[8+i] = bp[16+i];
    acc = __builtin_amdgcn_wmma_f32_16x16x32_f16(false, a, false, b,
                                                 (short)0, acc, false, false);
  }
  // stage relu(h) tile: C layout M = r + 8*hi, N = lane&15
  if (n < 8) {
    #pragma unroll
    for (int r = 0; r < 8; ++r) {
      float v = acc[r] + ((n < Hn) ? b1[n] : 0.f);
      hst[wave][r + 8*hi][n] = v > 0.f ? v : 0.f;
    }
  }
  __syncthreads();
  if (lane < 16) {
    int t = m0 + lane;
    float h[Hn];
    #pragma unroll
    for (int j = 0; j < Hn; ++j) h[j] = hst[wave][lane][j];
    #pragma unroll
    for (int l = 0; l < 4; ++l) {
      float m = bmu[l], lv = blv[l];
      #pragma unroll
      for (int j = 0; j < Hn; ++j) { m += h[j]*Wmu[j*4+l]; lv += h[j]*Wlv[j*4+l]; }
      mu_out[(size_t)t*4 + l]  = m;
      sig_out[(size_t)t*4 + l] = expf(0.5f*lv);
    }
  }
}

// ---------------- Kalman forward filter (sequential, 1 wave) --------------
__global__ __launch_bounds__(32) void fwd_kernel(
    const float* __restrict__ mu, const float* __restrict__ sig,
    const float* __restrict__ A,  const float* __restrict__ b_dyn,
    const float* __restrict__ Qp,
    float* __restrict__ fm, float* __restrict__ fP,
    float* __restrict__ pm, float* __restrict__ pP, float* __restrict__ ll_out)
{
  const int lane = threadIdx.x;
  float Af[16], bd[4], Q[16];
  #pragma unroll
  for (int i = 0; i < 16; ++i) Af[i] = A[i];
  #pragma unroll
  for (int i = 0; i < 4; ++i) bd[i] = b_dyn[i];
  { // Q = L L^T, diag(L) = exp(raw)
    float L[16];
    #pragma unroll
    for (int i = 0; i < 16; ++i) L[i] = 0.f;
    int idx = 0;
    #pragma unroll
    for (int i = 0; i < 4; ++i)
      #pragma unroll
      for (int j = 0; j < 4; ++j) if (j <= i) {
        float v = Qp[idx++];
        L[i*4+j] = (i == j) ? expf(v) : v;
      }
    mm4_bt(L, L, Q);
  }
  float m[4] = {0,0,0,0};
  float P[16] = {1,0,0,0, 0,1,0,0, 0,0,1,0, 0,0,0,1};
  float llsum = 0.f;
  const float LOG2PI = 1.8378770664093453f;

  for (int t = 0; t < Tn; ++t) {
    __builtin_prefetch(&mu[(size_t)(t+8)*4], 0, 1);
    __builtin_prefetch(&sig[(size_t)(t+8)*4], 0, 1);
    float y[4], r[4];
    #pragma unroll
    for (int l = 0; l < 4; ++l) { y[l] = mu[(size_t)t*4+l]; r[l] = sig[(size_t)t*4+l]; }
    float mp[4];
    #pragma unroll
    for (int i = 0; i < 4; ++i) {
      float s = bd[i];
      #pragma unroll
      for (int k = 0; k < 4; ++k) s += Af[i*4+k]*m[k];
      mp[i] = s;
    }
    float AP[16], Pp[16];
    mm4(Af, P, AP);
    mm4_bt(AP, Af, Pp);
    #pragma unroll
    for (int i = 0; i < 16; ++i) Pp[i] += Q[i];
    float S[16];
    #pragma unroll
    for (int i = 0; i < 16; ++i) S[i] = Pp[i];
    #pragma unroll
    for (int i = 0; i < 4; ++i) S[i*4+i] += r[i];
    float L[16], di[4];
    chol4(S, L, di);
    float K[16];                       // K = Pp S^{-1} = (S^{-1} Pp)^T
    cho_solve4_T(L, di, Pp, K);
    float v[4], mf[4];
    #pragma unroll
    for (int l = 0; l < 4; ++l) v[l] = y[l] - mp[l];
    #pragma unroll
    for (int i = 0; i < 4; ++i) {
      float s = mp[i];
      #pragma unroll
      for (int k = 0; k < 4; ++k) s += K[i*4+k]*v[k];
      mf[i] = s;
    }
    float IK[16], Pf[16];
    #pragma unroll
    for (int i = 0; i < 4; ++i)
      #pragma unroll
      for (int j = 0; j < 4; ++j) IK[i*4+j] = (i==j ? 1.f : 0.f) - K[i*4+j];
    mm4(IK, Pp, Pf);
    #pragma unroll
    for (int i = 0; i < 4; ++i)
      #pragma unroll
      for (int j = 0; j < 4; ++j) if (j < i) {
        float s = 0.5f*(Pf[i*4+j] + Pf[j*4+i]);
        Pf[i*4+j] = s; Pf[j*4+i] = s;
      }
    float w[4];
    #pragma unroll
    for (int i = 0; i < 4; ++i) {
      float s = v[i];
      #pragma unroll
      for (int k = 0; k < 4; ++k) if (k < i) s -= L[i*4+k]*w[k];
      w[i] = s * di[i];
    }
    float ww = w[0]*w[0]+w[1]*w[1]+w[2]*w[2]+w[3]*w[3];
    float ld = logf(L[0]) + logf(L[5]) + logf(L[10]) + logf(L[15]);
    llsum += -0.5f*(4.f*LOG2PI + ww) - ld;

    if (lane < 4)  { fm[(size_t)t*4+lane]  = mf[lane]; pm[(size_t)t*4+lane]  = mp[lane]; }
    if (lane < 16) { fP[(size_t)t*16+lane] = Pf[lane]; pP[(size_t)t*16+lane] = Pp[lane]; }
    #pragma unroll
    for (int i = 0; i < 4; ++i)  m[i] = mf[i];
    #pragma unroll
    for (int i = 0; i < 16; ++i) P[i] = Pf[i];
  }
  if (lane == 0) ll_out[0] = llsum;
}

// -------- RTS smoother + sample z + decoder layer-1 (sequential) ----------
__device__ __forceinline__ void emit_smoothed(
    int t, const float* ms, const float* Ps,
    const float* __restrict__ eps, const float* Wd1r, const float* bd1r,
    float* __restrict__ sm, float* __restrict__ sP,
    float* __restrict__ z_out, float* __restrict__ hd, int lane)
{
  if (lane < 4)  sm[(size_t)t*4+lane]  = ms[lane];
  if (lane < 16) sP[(size_t)t*16+lane] = Ps[lane];
  float Sp[16];
  #pragma unroll
  for (int i = 0; i < 16; ++i) Sp[i] = Ps[i];
  #pragma unroll
  for (int i = 0; i < 4; ++i) Sp[i*4+i] += 1e-6f;
  float L[16], di[4];
  chol4(Sp, L, di);
  float e[4], zv[4];
  #pragma unroll
  for (int l = 0; l < 4; ++l) e[l] = eps[(size_t)t*4+l];
  #pragma unroll
  for (int i = 0; i < 4; ++i) {
    float s = ms[i];
    #pragma unroll
    for (int k = 0; k < 4; ++k) if (k <= i) s += L[i*4+k]*e[k];
    zv[i] = s;
  }
  if (lane < 4) z_out[(size_t)t*4+lane] = zv[lane];
  if (lane < Hn) {
    float s = bd1r[lane];
    #pragma unroll
    for (int l = 0; l < 4; ++l) s += zv[l]*Wd1r[l*Hn+lane];
    hd[(size_t)t*Hn+lane] = s > 0.f ? s : 0.f;
  }
}

__global__ __launch_bounds__(32) void bwd_kernel(
    const float* __restrict__ fm, const float* __restrict__ fP,
    const float* __restrict__ pm, const float* __restrict__ pP,
    const float* __restrict__ A,  const float* __restrict__ eps,
    const float* __restrict__ Wd1, const float* __restrict__ bd1,
    float* __restrict__ sm, float* __restrict__ sP,
    float* __restrict__ z_out, float* __restrict__ hd)
{
  const int lane = threadIdx.x;
  float Af[16], Wd1r[4*Hn], bd1r[Hn];
  #pragma unroll
  for (int i = 0; i < 16; ++i) Af[i] = A[i];
  #pragma unroll
  for (int i = 0; i < 4*Hn; ++i) Wd1r[i] = Wd1[i];
  #pragma unroll
  for (int i = 0; i < Hn; ++i) bd1r[i] = bd1[i];

  float ms[4], Ps[16];
  #pragma unroll
  for (int i = 0; i < 4; ++i)  ms[i] = fm[(size_t)(Tn-1)*4+i];
  #pragma unroll
  for (int i = 0; i < 16; ++i) Ps[i] = fP[(size_t)(Tn-1)*16+i];
  emit_smoothed(Tn-1, ms, Ps, eps, Wd1r, bd1r, sm, sP, z_out, hd, lane);

  for (int t = Tn-2; t >= 0; --t) {
    __builtin_prefetch(&fP[(size_t)(t > 8 ? t-8 : 0)*16], 0, 1);
    __builtin_prefetch(&pP[(size_t)(t > 7 ? t-7 : 0)*16], 0, 1);
    float mf[4], Pf[16], mpn[4], Ppn[16];
    #pragma unroll
    for (int i = 0; i < 4; ++i)  { mf[i] = fm[(size_t)t*4+i]; mpn[i] = pm[(size_t)(t+1)*4+i]; }
    #pragma unroll
    for (int i = 0; i < 16; ++i) { Pf[i] = fP[(size_t)t*16+i]; Ppn[i] = pP[(size_t)(t+1)*16+i]; }
    float L[16], di[4];
    chol4(Ppn, L, di);
    float B[16], G[16];
    mm4(Af, Pf, B);                 // B = A P_f
    cho_solve4_T(L, di, B, G);      // G = P_f A^T Ppn^{-1}
    float dm[4], msn[4];
    #pragma unroll
    for (int i = 0; i < 4; ++i) dm[i] = ms[i] - mpn[i];
    #pragma unroll
    for (int i = 0; i < 4; ++i) {
      float s = mf[i];
      #pragma unroll
      for (int k = 0; k < 4; ++k) s += G[i*4+k]*dm[k];
      msn[i] = s;
    }
    float DP[16], GD[16], Psn[16];
    #pragma unroll
    for (int i = 0; i < 16; ++i) DP[i] = Ps[i] - Ppn[i];
    mm4(G, DP, GD);
    mm4_bt(GD, G, Psn);
    #pragma unroll
    for (int i = 0; i < 16; ++i) Psn[i] += Pf[i];
    #pragma unroll
    for (int i = 0; i < 4; ++i)
      #pragma unroll
      for (int j = 0; j < 4; ++j) if (j < i) {
        float s = 0.5f*(Psn[i*4+j] + Psn[j*4+i]);
        Psn[i*4+j] = s; Psn[j*4+i] = s;
      }
    #pragma unroll
    for (int i = 0; i < 4; ++i)  ms[i] = msn[i];
    #pragma unroll
    for (int i = 0; i < 16; ++i) Ps[i] = Psn[i];
    emit_smoothed(t, ms, Ps, eps, Wd1r, bd1r, sm, sP, z_out, hd, lane);
  }
}

// ---------------- Decoder GEMM: x_recon = hd @ Wd2 + bd2 ------------------
__global__ __launch_bounds__(128) void dec_kernel(
    const float* __restrict__ hd, const float* __restrict__ Wd2,
    const float* __restrict__ bd2, float* __restrict__ xrec)
{
  // Transposed, K-padded f16 image of Wd2: w2t[d][k], k padded 6->32 (zeros)
  // so both lane-halves build B fragments branch-free; row stride 40 halves
  // (20 dwords) keeps the 16 N-lanes on distinct banks.
  __shared__ _Float16 w2t[Dn][W2K];
  const int tid = threadIdx.x;
  for (int idx = tid; idx < Dn*W2K; idx += 128) {
    int d = idx / W2K, k = idx % W2K;
    w2t[d][k] = (_Float16)((k < Hn) ? Wd2[(size_t)k*Dn + d] : 0.f);
  }
  __syncthreads();

  const int wave = tid >> 5, lane = tid & 31;
  const int m0 = (blockIdx.x*4 + wave) * 16;
  const int hi = lane >> 4;
  const int n  = lane & 15;

  v16h a;                               // K=6 padded into K=32 frag
  #pragma unroll
  for (int i = 0; i < 16; ++i) a[i] = (_Float16)0.f;
  if (hi == 0) {                        // K=16..31 rows are zero
    const float* hp = hd + (size_t)(m0 + n)*Hn;
    #pragma unroll
    for (int j = 0; j < Hn; ++j) a[j] = (_Float16)hp[j];
  }

  for (int n0 = 0; n0 < Dn; n0 += 16) {
    const _Float16* bp = &w2t[n0 + n][hi*16];
    v16h b;
    #pragma unroll
    for (int i = 0; i < 16; ++i) b[i] = bp[i];
    v8f c = {};
    c = __builtin_amdgcn_wmma_f32_16x16x32_f16(false, a, false, b,
                                               (short)0, c, false, false);
    float bias = bd2[n0 + n];
    #pragma unroll
    for (int r = 0; r < 8; ++r)
      xrec[(size_t)(m0 + r + 8*hi)*Dn + n0 + n] = c[r] + bias;
  }
}

// --------------------------------- launch ---------------------------------
extern "C" void kernel_launch(void* const* d_in, const int* in_sizes, int n_in,
                              void* d_out, int out_size, void* d_ws, size_t ws_size,
                              hipStream_t stream) {
  const float* x    = (const float*)d_in[0];
  const float* eps  = (const float*)d_in[1];
  const float* W1   = (const float*)d_in[2];
  const float* b1   = (const float*)d_in[3];
  const float* Wmu  = (const float*)d_in[4];
  const float* bmu  = (const float*)d_in[5];
  const float* Wlv  = (const float*)d_in[6];
  const float* blv  = (const float*)d_in[7];
  const float* Wd1  = (const float*)d_in[8];
  const float* bd1  = (const float*)d_in[9];
  const float* Wd2  = (const float*)d_in[10];
  const float* bd2  = (const float*)d_in[11];
  const float* A    = (const float*)d_in[12];
  const float* bdyn = (const float*)d_in[13];
  const float* Qp   = (const float*)d_in[14];

  float* out = (float*)d_out;
  size_t o = 0;
  float* xrec = out + o; o += (size_t)Tn*Dn;
  float* zo   = out + o; o += (size_t)Tn*4;
  float* muo  = out + o; o += (size_t)Tn*4;
  float* sigo = out + o; o += (size_t)Tn*4;
  float* fmo  = out + o; o += (size_t)Tn*4;
  float* fPo  = out + o; o += (size_t)Tn*16;
  float* smo  = out + o; o += (size_t)Tn*4;
  float* sPo  = out + o; o += (size_t)Tn*16;
  float* pmo  = out + o; o += (size_t)Tn*4;
  float* pPo  = out + o; o += (size_t)Tn*16;
  float* llo  = out + o;
  float* hd   = (float*)d_ws;          // [Tn, 6] fp32 scratch (768 KiB)

  enc_kernel<<<Tn/64, 128, 0, stream>>>(x, W1, b1, Wmu, bmu, Wlv, blv, muo, sigo);
  fwd_kernel<<<1, 32, 0, stream>>>(muo, sigo, A, bdyn, Qp, fmo, fPo, pmo, pPo, llo);
  bwd_kernel<<<1, 32, 0, stream>>>(fmo, fPo, pmo, pPo, A, eps, Wd1, bd1, smo, sPo, zo, hd);
  dec_kernel<<<Tn/64, 128, 0, stream>>>(hd, Wd2, bd2, xrec);
}
